// SyntaxBertSelfAttention_90099823935721
// MI455X (gfx1250) — matrix operand order repaired and checked by reading, not compiled, not run on hardware
//
#include <hip/hip_runtime.h>
#include <hip/hip_bf16.h>
#include <math.h>

// ---------------------------------------------------------------------------
// BERT self-attention for MI455X (gfx1250, wave32, WMMA)
// B=2, S=2048, D=768, H=12, HD=64, scale = 1/sqrt(12)
// ---------------------------------------------------------------------------

typedef __attribute__((ext_vector_type(16))) _Float16 v16h;
typedef __attribute__((ext_vector_type(8)))  _Float16 v8h;
typedef __attribute__((ext_vector_type(4)))  _Float16 v4h;
typedef __attribute__((ext_vector_type(8)))  float    v8f;

#define NB   2
#define NS   2048
#define ND   768
#define NH   12
#define NHD  64
#define NBH  (NB * NH)          // 24
#define SCALE_F 0.28867513459481287f   // 1/sqrt(12)

union V16U { v16h v; v8h h[2]; };

// Build a 16-half WMMA operand from two 16-byte chunks (per-lane addresses).
__device__ inline v16h ld_pair(const _Float16* p0, const _Float16* p1) {
    V16U r;
    r.h[0] = *(const v8h*)p0;
    r.h[1] = *(const v8h*)p1;
    return r.v;
}

__device__ inline v8f wmma_f16(v16h a, v16h b, v8f c) {
    return __builtin_amdgcn_wmma_f32_16x16x32_f16(
        /*neg_a=*/false, a, /*neg_b=*/false, b,
        /*c_mod=*/(short)0, c, /*reuse_a=*/false, /*reuse_b=*/false);
}

// ---------------------------------------------------------------------------
// Kernel 1: hidden f32 -> f16, vectorized x4.
// ---------------------------------------------------------------------------
__global__ __launch_bounds__(256) void cvt_hidden(const float* __restrict__ x,
                                                  _Float16* __restrict__ y) {
    int i = blockIdx.x * 256 + threadIdx.x;           // grid sized exactly
    const float4 f = ((const float4*)x)[i];
    v4h o;
    o.x = (_Float16)f.x; o.y = (_Float16)f.y;
    o.z = (_Float16)f.z; o.w = (_Float16)f.w;
    *(v4h*)(y + 4 * i) = o;
}

// ---------------------------------------------------------------------------
// Kernel 2: transpose + convert weights: Wt[mat][out][in] = W[mat][in][out]
// grid: (3, 768, 3) x 256 threads
// ---------------------------------------------------------------------------
__global__ __launch_bounds__(256) void transpose_w(const float* __restrict__ Wq,
                                                   const float* __restrict__ Wk,
                                                   const float* __restrict__ Wv,
                                                   _Float16* __restrict__ Wt) {
    const int i   = blockIdx.x * 256 + threadIdx.x;   // input dim 0..767
    const int o   = blockIdx.y;                       // output dim 0..767
    const int mat = blockIdx.z;                       // 0=q,1=k,2=v
    const float* W = (mat == 0) ? Wq : (mat == 1) ? Wk : Wv;
    Wt[(size_t)mat * ND * ND + (size_t)o * ND + i] = (_Float16)W[(size_t)i * ND + o];
}

// ---------------------------------------------------------------------------
// Kernel 3: QKV projection. One 16x64 output strip per wave (one head's cols).
// `mat` comes from blockIdx.z (true scalar -> SALU selects, no v_cndmask).
// k-loop is manually double-buffered: next iteration's A/B operands are in
// flight while the current WMMAs execute -> partial loadcnt waits.
// Q,K stored [BH][S][64] f16 ; V stored transposed [BH][64][S] f16.
// grid: (384, 1, 3) x 256 threads ; 3072 waves per mat
// ---------------------------------------------------------------------------
__global__ __launch_bounds__(256) void qkv_gemm(const _Float16* __restrict__ Xh,  // [4096][768]
                                                const _Float16* __restrict__ Wt,  // [3][768][768]
                                                const float* __restrict__ bq,
                                                const float* __restrict__ bk,
                                                const float* __restrict__ bv,
                                                _Float16* __restrict__ Qb,
                                                _Float16* __restrict__ Kb,
                                                _Float16* __restrict__ Vtb) {
    const int lane = threadIdx.x & 31;
    const int wave = threadIdx.x >> 5;
    const int wid  = blockIdx.x * 8 + wave;   // 0..3071
    const int mat  = blockIdx.z;              // scalar: 0=q,1=k,2=v

    const int rt = wid & 255;                 // row tile 0..255
    const int cg = wid >> 8;                  // head / 64-col group 0..11

    const int nl   = lane & 15;               // A row within tile == B col within tile
    const int half = lane >> 4;

    const _Float16* a_base = Xh + (size_t)(rt * 16 + nl) * ND;
    const _Float16* b_base = Wt + (size_t)mat * ND * ND + (size_t)(cg * 64 + nl) * ND;

    v8f acc[4] = {{}, {}, {}, {}};

    // prologue: load k=0 operands
    v16h a_cur = ld_pair(a_base + half * 16, a_base + half * 16 + 8);
    v16h b_cur[4];
    #pragma unroll
    for (int d = 0; d < 4; ++d) {
        const _Float16* bp = b_base + (size_t)d * 16 * ND + half * 16;
        b_cur[d] = ld_pair(bp, bp + 8);
    }

    #pragma unroll 2
    for (int k = 32; k < ND; k += 32) {
        // issue next iteration's loads first
        v16h a_nxt = ld_pair(a_base + k + half * 16, a_base + k + half * 16 + 8);
        v16h b_nxt[4];
        #pragma unroll
        for (int d = 0; d < 4; ++d) {
            const _Float16* bp = b_base + (size_t)d * 16 * ND + k + half * 16;
            b_nxt[d] = ld_pair(bp, bp + 8);
        }
        // consume current operands
        #pragma unroll
        for (int d = 0; d < 4; ++d) acc[d] = wmma_f16(a_cur, b_cur[d], acc[d]);
        a_cur = a_nxt;
        #pragma unroll
        for (int d = 0; d < 4; ++d) b_cur[d] = b_nxt[d];
    }
    #pragma unroll
    for (int d = 0; d < 4; ++d) acc[d] = wmma_f16(a_cur, b_cur[d], acc[d]);

    // bias (scalar select on mat)
    const float* bias;
    if (mat == 0)      bias = bq;
    else if (mat == 1) bias = bk;
    else               bias = bv;
    #pragma unroll
    for (int d = 0; d < 4; ++d) {
        const float bvv = bias[cg * 64 + d * 16 + nl];
        #pragma unroll
        for (int r = 0; r < 8; ++r) acc[d][r] += bvv;
    }

    // C layout: element r on this lane is (row = r + 8*half, col = d*16 + nl).
    if (mat < 2) {
        _Float16* dst = (mat == 0) ? Qb : Kb;     // scalar select
        #pragma unroll
        for (int r = 0; r < 8; ++r) {
            const int row = rt * 16 + r + half * 8;   // 0..4095
            const int bb  = row >> 11;                // batch
            const int ss  = row & 2047;               // seq
            _Float16* p = dst + (((size_t)(bb * NH + cg)) * NS + ss) * NHD + nl;
            #pragma unroll
            for (int d = 0; d < 4; ++d) p[d * 16] = (_Float16)acc[d][r];
        }
    } else {
        #pragma unroll
        for (int r = 0; r < 8; ++r) {
            const int row = rt * 16 + r + half * 8;
            const int bb  = row >> 11;
            const int ss  = row & 2047;
            _Float16* p = Vtb + (((size_t)(bb * NH + cg)) * NHD + nl) * NS + ss;
            #pragma unroll
            for (int d = 0; d < 4; ++d) p[(size_t)d * 16 * NS] = (_Float16)acc[d][r];
        }
    }
}

// ---------------------------------------------------------------------------
// Kernel 4: flash attention. One wave per (bh, 16-row query tile); 64-key
// steps. Softmax denominator via extra WMMAs (P x ones) -> no sum shuffles.
// K operands double-buffered across iterations (branchless wrap on the final
// prefetch); V loads issued before the softmax VALU block.
// 3072 waves total; 4 waves/block -> 768 blocks. LDS: 16x64 f16 per wave.
// ---------------------------------------------------------------------------
__global__ __launch_bounds__(128) void attn(const _Float16* __restrict__ Qb,   // [24][2048][64]
                                            const _Float16* __restrict__ Kb,   // [24][2048][64]
                                            const _Float16* __restrict__ Vtb,  // [24][64][2048]
                                            const float* __restrict__ mask,    // [2][2048]
                                            float* __restrict__ out) {         // [2][2048][768]
    __shared__ _Float16 lds[4][16 * 64];

    const int lane = threadIdx.x & 31;
    const int wave = threadIdx.x >> 5;
    const int wid  = blockIdx.x * 4 + wave;
    const int bh   = wid >> 7;              // 0..23
    const int qt   = wid & 127;             // query tile 0..127
    const int b    = bh / NH;
    const int h    = bh % NH;

    const int nl   = lane & 15;
    const int half = lane >> 4;

    // Q tile as two A-operands (K-dim chunks 0..31, 32..63); stays in VGPRs.
    const _Float16* Qp = Qb + ((size_t)bh * NS + qt * 16 + nl) * NHD;
    const v16h qa0 = ld_pair(Qp + half * 16,      Qp + half * 16 + 8);
    const v16h qa1 = ld_pair(Qp + 32 + half * 16, Qp + 40 + half * 16);

    const _Float16* Kp = Kb  + (size_t)bh * NS * NHD;
    const _Float16* Vp = Vtb + (size_t)bh * NHD * NS;
    const float*    mp = mask + (size_t)b * NS;

    // o[0..3] = output chunks (cols d*16..d*16+15); o[4] = row-sum accumulator
    v8f o[5] = {{}, {}, {}, {}, {}};
    float mrow[8];
    #pragma unroll
    for (int r = 0; r < 8; ++r) mrow[r] = -1e30f;

    // ones B-operand for the row-sum WMMA (every element 1.0)
    v16h ones;
    #pragma unroll
    for (int i = 0; i < 16; ++i) ones[i] = (_Float16)1.0f;

    _Float16* P = &lds[wave][0];

    // K-operand loader: 4 key tiles x 2 kd chunks for a 64-key step
    auto loadK = [&](int jb, v16h* kb) {
        #pragma unroll
        for (int t = 0; t < 4; ++t) {
            const _Float16* kt = Kp + (size_t)(jb + 16 * t + nl) * NHD + half * 16;
            kb[2 * t]     = ld_pair(kt,      kt + 8);
            kb[2 * t + 1] = ld_pair(kt + 32, kt + 40);
        }
    };

    v16h kb[8];
    loadK(0, kb);

    #pragma unroll 2
    for (int jb = 0; jb < NS; jb += 64) {
        // ---- scores: 4 tiles of 16 keys each (8 WMMAs on prefetched kb) ----
        v8f s[4];
        #pragma unroll
        for (int t = 0; t < 4; ++t) {
            v8f ss = {};
            ss = wmma_f16(qa0, kb[2 * t],     ss);
            ss = wmma_f16(qa1, kb[2 * t + 1], ss);
            const float mk = mp[jb + 16 * t + nl];
            #pragma unroll
            for (int r = 0; r < 8; ++r) s[t][r] = ss[r] * SCALE_F + mk;
        }

        // ---- issue V loads + next-step K prefetch: latency hides under softmax ----
        v16h vb[8];
        #pragma unroll
        for (int d = 0; d < 4; ++d) {
            const _Float16* vp = Vp + (size_t)(d * 16 + nl) * NS + jb + half * 16;
            vb[2 * d]     = ld_pair(vp,      vp + 8);
            vb[2 * d + 1] = ld_pair(vp + 32, vp + 40);
        }
        v16h kn[8];
        loadK((jb + 64) & (NS - 1), kn);    // branchless wrap; last prefetch unused

        // ---- online max (only reduction needing shuffles) ----
        float fs[8];
        #pragma unroll
        for (int r = 0; r < 8; ++r) {
            float v = fmaxf(fmaxf(s[0][r], s[1][r]), fmaxf(s[2][r], s[3][r]));
            v = fmaxf(v, __shfl_xor(v, 1, 32));
            v = fmaxf(v, __shfl_xor(v, 2, 32));
            v = fmaxf(v, __shfl_xor(v, 4, 32));
            v = fmaxf(v, __shfl_xor(v, 8, 32));
            const float mnew = fmaxf(mrow[r], v);
            fs[r]   = __expf(mrow[r] - mnew);
            mrow[r] = mnew;
        }

        // ---- P = exp(s - m) staged to LDS (C-layout -> row-major memory) ----
        #pragma unroll
        for (int r = 0; r < 8; ++r) {
            const int row = r + half * 8;
            #pragma unroll
            for (int t = 0; t < 4; ++t)
                P[row * 64 + 16 * t + nl] = (_Float16)__expf(s[t][r] - mrow[r]);
        }

        // ---- rescale all running accumulators (incl. row-sum) ----
        #pragma unroll
        for (int r = 0; r < 8; ++r) {
            o[0][r] *= fs[r]; o[1][r] *= fs[r]; o[2][r] *= fs[r];
            o[3][r] *= fs[r]; o[4][r] *= fs[r];
        }

        // ---- reload P in A-layout: two 16x32 operands (keys jb..+31, +32..+63) ----
        const v16h pa0 = ld_pair(P + nl * 64 + half * 16,      P + nl * 64 + half * 16 + 8);
        const v16h pa1 = ld_pair(P + nl * 64 + 32 + half * 16, P + nl * 64 + 40 + half * 16);

        // ---- O += P x V : 8 WMMAs; row-sum += P x ones : 2 WMMAs ----
        #pragma unroll
        for (int d = 0; d < 4; ++d) {
            o[d] = wmma_f16(pa0, vb[2 * d],     o[d]);
            o[d] = wmma_f16(pa1, vb[2 * d + 1], o[d]);
        }
        o[4] = wmma_f16(pa0, ones, o[4]);
        o[4] = wmma_f16(pa1, ones, o[4]);

        // ---- rotate K double-buffer (renamed away under unroll 2) ----
        #pragma unroll
        for (int i = 0; i < 8; ++i) kb[i] = kn[i];
    }

    // ---- normalize by row-sum and store f32 output [B][S][768] ----
    #pragma unroll
    for (int r = 0; r < 8; ++r) {
        const int row = qt * 16 + r + half * 8;
        const float inv = 1.0f / o[4][r];
        float* op = out + ((size_t)b * NS + row) * ND + h * NHD + nl;
        op[0]  = o[0][r] * inv;
        op[16] = o[1][r] * inv;
        op[32] = o[2][r] * inv;
        op[48] = o[3][r] * inv;
    }
}

// ---------------------------------------------------------------------------
// Launch
// ---------------------------------------------------------------------------
extern "C" void kernel_launch(void* const* d_in, const int* in_sizes, int n_in,
                              void* d_out, int out_size, void* d_ws, size_t ws_size,
                              hipStream_t stream) {
    const float* hidden = (const float*)d_in[0];
    const float* amask  = (const float*)d_in[1];
    const float* Wq     = (const float*)d_in[2];
    const float* bq     = (const float*)d_in[3];
    const float* Wk     = (const float*)d_in[4];
    const float* bk     = (const float*)d_in[5];
    const float* Wv     = (const float*)d_in[6];
    const float* bv     = (const float*)d_in[7];
    float* out          = (float*)d_out;

    // workspace carve (all f16)
    char* ws = (char*)d_ws;
    _Float16* Xh  = (_Float16*)ws;  ws += (size_t)NB * NS * ND * 2;        // 6.0 MB
    _Float16* Wt  = (_Float16*)ws;  ws += (size_t)3 * ND * ND * 2;         // 3.4 MB
    _Float16* Qb  = (_Float16*)ws;  ws += (size_t)NBH * NS * NHD * 2;      // 6.0 MB
    _Float16* Kb  = (_Float16*)ws;  ws += (size_t)NBH * NS * NHD * 2;      // 6.0 MB
    _Float16* Vtb = (_Float16*)ws;                                         // 6.0 MB

    // 1) hidden -> f16  (B*S*D/4 = 786432 threads)
    cvt_hidden<<<dim3(3072), dim3(256), 0, stream>>>(hidden, Xh);

    // 2) weights -> f16 transposed
    transpose_w<<<dim3(3, ND, 3), dim3(256), 0, stream>>>(Wq, Wk, Wv, Wt);

    // 3) QKV projection: 3072 strip-waves per mat, 8 waves/block
    qkv_gemm<<<dim3(384, 1, 3), dim3(256), 0, stream>>>(Xh, Wt, bq, bk, bv, Qb, Kb, Vtb);

    // 4) flash attention: 24*128 = 3072 waves, 4 waves/block -> 768 blocks
    attn<<<dim3(768), dim3(128), 0, stream>>>(Qb, Kb, Vtb, amask, out);
}
